// InferenceLTMBlock_43087111913967
// MI455X (gfx1250) — matrix operand order repaired and checked by reading, not compiled, not run on hardware
//
#include <hip/hip_runtime.h>
#include <math.h>

typedef __attribute__((ext_vector_type(16))) _Float16 v16h;
typedef __attribute__((ext_vector_type(8)))  _Float16 v8h;
typedef __attribute__((ext_vector_type(8)))  float    v8f;

union Frag16 { v16h v; v8h h[2]; };

__device__ __forceinline__ v8h ld_v8h(const _Float16* p) {
  return *reinterpret_cast<const v8h*>(p);
}
__device__ __forceinline__ void st_v8h(_Float16* p, v8h x) {
  *reinterpret_cast<v8h*>(p) = x;
}

// CDNA5 async global->LDS copy (16B), tracked by ASYNCcnt.
// LDS byte offset = low 32 bits of the generic pointer (LDS aperture keeps
// the offset in addr[31:0]).
__device__ __forceinline__ void async_ld_b128(_Float16* lds_dst, const _Float16* gsrc) {
  unsigned int loff = (unsigned int)(unsigned long long)lds_dst;
  asm volatile("global_load_async_to_lds_b128 %0, %1, off"
               :: "v"(loff), "v"(gsrc) : "memory");
}
__device__ __forceinline__ void wait_async0() {
  asm volatile("s_wait_asynccnt 0" ::: "memory");
}

#define C_DIM 1024
#define HEADS 16
#define DHEAD 64

// ---------------------------------------------------------------------------
// f32[K][N] -> f16[N][K] transposed weight conversion (one-time, 32x32 tiles)
// ---------------------------------------------------------------------------
__global__ __launch_bounds__(256) void k_transpose_f16(
    const float* __restrict__ in, _Float16* __restrict__ out, int K, int N) {
  __shared__ float tile[32][33];
  const int bx = blockIdx.x * 32;   // N
  const int by = blockIdx.y * 32;   // K
  const int tx = threadIdx.x & 31;
  const int ty = threadIdx.x >> 5;  // 8 rows per pass
#pragma unroll
  for (int r = ty; r < 32; r += 8)
    tile[r][tx] = in[(size_t)(by + r) * N + bx + tx];
  __syncthreads();
#pragma unroll
  for (int r = ty; r < 32; r += 8)
    out[(size_t)(bx + r) * K + by + tx] = (_Float16)tile[tx][r];
}

// ---------------------------------------------------------------------------
// log(size)
// ---------------------------------------------------------------------------
__global__ void k_logsize(const float* __restrict__ sz, float* __restrict__ out, int n) {
  int i = blockIdx.x * blockDim.x + threadIdx.x;
  if (i < n) out[i] = logf(sz[i]);
}

// ---------------------------------------------------------------------------
// LayerNorm (one block per row), f32 in -> f16 out
// ---------------------------------------------------------------------------
__global__ __launch_bounds__(256) void k_layernorm_f16(
    const float* __restrict__ x, const float* __restrict__ g, const float* __restrict__ b,
    _Float16* __restrict__ out, int C) {
  const int row = blockIdx.x;
  const float* xr = x + (size_t)row * C;
  float s = 0.f, ss = 0.f;
  for (int c = threadIdx.x; c < C; c += 256) { float v = xr[c]; s += v; ss += v * v; }
  for (int m = 16; m >= 1; m >>= 1) { s += __shfl_xor(s, m, 32); ss += __shfl_xor(ss, m, 32); }
  __shared__ float rs[8], rss[8];
  __shared__ float mu_s, rstd_s;
  if ((threadIdx.x & 31) == 0) { rs[threadIdx.x >> 5] = s; rss[threadIdx.x >> 5] = ss; }
  __syncthreads();
  if (threadIdx.x == 0) {
    float S = 0.f, SS = 0.f;
    for (int i = 0; i < 8; ++i) { S += rs[i]; SS += rss[i]; }
    float mu = S / (float)C;
    float var = SS / (float)C - mu * mu;
    mu_s = mu; rstd_s = rsqrtf(var + 1e-5f);
  }
  __syncthreads();
  const float mu = mu_s, rstd = rstd_s;
  for (int c = threadIdx.x; c < C; c += 256)
    out[(size_t)row * C + c] = (_Float16)((xr[c] - mu) * rstd * g[c] + b[c]);
}

// ---------------------------------------------------------------------------
// WMMA GEMM: C[M,N] = A[M,K](f16) @ Bt[N,K](f16, pre-transposed)
// fused bias / exact GELU / residual. 128x128x32 tiles, async LDS fills.
// ---------------------------------------------------------------------------
#define TM 128
#define TN 128
#define TK 32
#define LPAD 8

__global__ __launch_bounds__(256) void k_gemm_f16(
    const _Float16* __restrict__ A, const _Float16* __restrict__ Bt,
    const float* __restrict__ bias, const float* __restrict__ residual,
    float* __restrict__ outF, _Float16* __restrict__ outH,
    int M, int N, int K, int act) {
  __shared__ _Float16 lA[TM][TK + LPAD];   // [m][k]
  __shared__ _Float16 lB[TN][TK + LPAD];   // [n][k]

  const int tid  = threadIdx.x;
  const int lane = tid & 31;
  const int w    = tid >> 5;     // 0..7
  const int wm   = w >> 1;       // 0..3
  const int wn   = w & 1;        // 0..1
  const int hi   = lane >> 4;    // lane half
  const int nl   = lane & 15;
  const int m0   = blockIdx.y * TM;
  const int n0   = blockIdx.x * TN;

  v8f acc[2][4];
#pragma unroll
  for (int i = 0; i < 2; ++i)
#pragma unroll
    for (int j = 0; j < 4; ++j)
#pragma unroll
      for (int r = 0; r < 8; ++r) acc[i][j][r] = 0.f;

  const int lrow  = tid >> 1;           // 0..127 (both A rows and B cols)
  const int lpart = (tid & 1) * 16;     // 0 / 16

  const int agr = m0 + lrow;
  const _Float16* aptr = A  + (size_t)agr * K + lpart;          // + k0
  const _Float16* bptr = Bt + (size_t)(n0 + lrow) * K + lpart;  // + k0
  const bool avalid = (agr < M);

  for (int k0 = 0; k0 < K; k0 += TK) {
    if (avalid) {
      async_ld_b128(&lA[lrow][lpart],     aptr + k0);
      async_ld_b128(&lA[lrow][lpart + 8], aptr + k0 + 8);
    }
    async_ld_b128(&lB[lrow][lpart],     bptr + k0);
    async_ld_b128(&lB[lrow][lpart + 8], bptr + k0 + 8);
    if (k0 + TK < K) {   // prefetch next tile into L2
      __builtin_prefetch(aptr + k0 + TK, 0, 1);
      __builtin_prefetch(bptr + k0 + TK, 0, 1);
    }
    wait_async0();
    __syncthreads();

    Frag16 af[2], bf[4];
#pragma unroll
    for (int i = 0; i < 2; ++i) {
      const _Float16* base = &lA[wm * 32 + i * 16 + nl][0];
      af[i].h[0] = ld_v8h(base + hi * 8);
      af[i].h[1] = ld_v8h(base + 16 + hi * 8);
    }
#pragma unroll
    for (int j = 0; j < 4; ++j) {
      const _Float16* base = &lB[wn * 64 + j * 16 + nl][0];
      bf[j].h[0] = ld_v8h(base + hi * 8);
      bf[j].h[1] = ld_v8h(base + 16 + hi * 8);
    }
#pragma unroll
    for (int i = 0; i < 2; ++i)
#pragma unroll
      for (int j = 0; j < 4; ++j)
        acc[i][j] = __builtin_amdgcn_wmma_f32_16x16x32_f16(
            false, af[i].v, false, bf[j].v, (short)0, acc[i][j], false, false);
    __syncthreads();
  }

#pragma unroll
  for (int i = 0; i < 2; ++i)
#pragma unroll
    for (int j = 0; j < 4; ++j) {
      const int col = n0 + wn * 64 + j * 16 + nl;
#pragma unroll
      for (int r = 0; r < 8; ++r) {
        const int row = m0 + wm * 32 + i * 16 + hi * 8 + r;
        if (row < M) {
          float v = acc[i][j][r];
          if (bias)     v += bias[col];
          if (act == 1) v = 0.5f * v * (1.0f + erff(v * 0.70710678118654752f));
          if (residual) v += residual[(size_t)row * N + col];
          if (outF) outF[(size_t)row * N + col] = v;
          if (outH) outH[(size_t)row * N + col] = (_Float16)v;
        }
      }
    }
}

// ---------------------------------------------------------------------------
// Flash attention (wave32 WMMA): qkv layout [n][3*1024], head h slice h*64
// grid = (N/128, HEADS), block 256 (8 waves, 16 queries/wave)
// ---------------------------------------------------------------------------
__global__ __launch_bounds__(256) void k_attn_flash(
    const _Float16* __restrict__ qkv, const float* __restrict__ logsz,
    _Float16* __restrict__ outH, int Ntok) {
  __shared__ _Float16 sK[32][72];        // [key][d]
  __shared__ _Float16 sVt[64][40];       // [d][key]  (transposed)
  __shared__ _Float16 sP[8][16][40];     // per-wave P scratch [q][key]

  const int tid  = threadIdx.x;
  const int lane = tid & 31;
  const int w    = tid >> 5;
  const int hi   = lane >> 4;
  const int nl   = lane & 15;
  const int h    = blockIdx.y;
  const int q0   = blockIdx.x * 128;
  const int qrow = q0 + w * 16 + nl;

  Frag16 aq[2];
  {
    const _Float16* qp = qkv + (size_t)qrow * (3 * C_DIM) + h * DHEAD;
    aq[0].h[0] = ld_v8h(qp + hi * 8);
    aq[0].h[1] = ld_v8h(qp + 16 + hi * 8);
    aq[1].h[0] = ld_v8h(qp + 32 + hi * 8);
    aq[1].h[1] = ld_v8h(qp + 48 + hi * 8);
  }

  v8f o[4];
  float mrun[8], lrun[8];
#pragma unroll
  for (int j = 0; j < 4; ++j)
#pragma unroll
    for (int r = 0; r < 8; ++r) o[j][r] = 0.f;
#pragma unroll
  for (int r = 0; r < 8; ++r) { mrun[r] = -1e30f; lrun[r] = 0.f; }

  const int lkey = tid >> 3;        // 0..31
  const int ld0  = (tid & 7) * 8;   // 0..56

  for (int kt = 0; kt < Ntok; kt += 32) {
    {
      // K tile: direct async copy into LDS
      const _Float16* kp = qkv + (size_t)(kt + lkey) * (3 * C_DIM) + C_DIM + h * DHEAD + ld0;
      async_ld_b128(&sK[lkey][ld0], kp);
      // V tile: needs transpose -> regular load + ds stores
      const _Float16* vp = qkv + (size_t)(kt + lkey) * (3 * C_DIM) + 2 * C_DIM + h * DHEAD + ld0;
      v8h vv = ld_v8h(vp);
#pragma unroll
      for (int j = 0; j < 8; ++j) sVt[ld0 + j][lkey] = vv[j];
    }
    wait_async0();
    __syncthreads();

    // S = Q @ K^T  (two key sub-tiles of 16)
    Frag16 b0a, b0b, b1a, b1b;
    {
      const _Float16* kb0 = &sK[nl][0];
      const _Float16* kb1 = &sK[16 + nl][0];
      b0a.h[0] = ld_v8h(kb0 + hi * 8);      b0a.h[1] = ld_v8h(kb0 + 16 + hi * 8);
      b0b.h[0] = ld_v8h(kb0 + 32 + hi * 8); b0b.h[1] = ld_v8h(kb0 + 48 + hi * 8);
      b1a.h[0] = ld_v8h(kb1 + hi * 8);      b1a.h[1] = ld_v8h(kb1 + 16 + hi * 8);
      b1b.h[0] = ld_v8h(kb1 + 32 + hi * 8); b1b.h[1] = ld_v8h(kb1 + 48 + hi * 8);
    }
    v8f z;
#pragma unroll
    for (int r = 0; r < 8; ++r) z[r] = 0.f;
    v8f s0 = __builtin_amdgcn_wmma_f32_16x16x32_f16(false, aq[0].v, false, b0a.v, (short)0, z,  false, false);
    s0     = __builtin_amdgcn_wmma_f32_16x16x32_f16(false, aq[1].v, false, b0b.v, (short)0, s0, false, false);
    v8f s1 = __builtin_amdgcn_wmma_f32_16x16x32_f16(false, aq[0].v, false, b1a.v, (short)0, z,  false, false);
    s1     = __builtin_amdgcn_wmma_f32_16x16x32_f16(false, aq[1].v, false, b1b.v, (short)0, s1, false, false);

    const float bias0 = logsz[kt + nl];
    const float bias1 = logsz[kt + 16 + nl];
    float scl[8];
#pragma unroll
    for (int r = 0; r < 8; ++r) {
      float v0 = s0[r] * 0.125f + bias0;   // dh^-0.5 = 1/8
      float v1 = s1[r] * 0.125f + bias1;
      float mx = fmaxf(v0, v1);
      mx = fmaxf(mx, __shfl_xor(mx, 1, 32));
      mx = fmaxf(mx, __shfl_xor(mx, 2, 32));
      mx = fmaxf(mx, __shfl_xor(mx, 4, 32));
      mx = fmaxf(mx, __shfl_xor(mx, 8, 32));
      float mnew = fmaxf(mrun[r], mx);
      scl[r] = expf(mrun[r] - mnew);
      float p0 = expf(v0 - mnew);
      float p1 = expf(v1 - mnew);
      float rsum = p0 + p1;
      rsum += __shfl_xor(rsum, 1, 32);
      rsum += __shfl_xor(rsum, 2, 32);
      rsum += __shfl_xor(rsum, 4, 32);
      rsum += __shfl_xor(rsum, 8, 32);
      lrun[r] = lrun[r] * scl[r] + rsum;
      mrun[r] = mnew;
      sP[w][8 * hi + r][nl]      = (_Float16)p0;
      sP[w][8 * hi + r][16 + nl] = (_Float16)p1;
    }
#pragma unroll
    for (int j = 0; j < 4; ++j)
#pragma unroll
      for (int r = 0; r < 8; ++r) o[j][r] *= scl[r];

    // P (16x32) @ V (32x64): A-frag from per-wave LDS (same-wave LDS ordered)
    Frag16 ap;
    {
      const _Float16* pb = &sP[w][nl][0];
      ap.h[0] = ld_v8h(pb + hi * 8);
      ap.h[1] = ld_v8h(pb + 16 + hi * 8);
    }
#pragma unroll
    for (int j = 0; j < 4; ++j) {
      const _Float16* vb = &sVt[j * 16 + nl][0];
      Frag16 bv;
      bv.h[0] = ld_v8h(vb + hi * 8);
      bv.h[1] = ld_v8h(vb + 16 + hi * 8);
      o[j] = __builtin_amdgcn_wmma_f32_16x16x32_f16(false, ap.v, false, bv.v, (short)0, o[j], false, false);
    }
    __syncthreads();
  }

#pragma unroll
  for (int j = 0; j < 4; ++j)
#pragma unroll
    for (int r = 0; r < 8; ++r) {
      const int q = q0 + w * 16 + 8 * hi + r;
      const int d = j * 16 + nl;
      outH[(size_t)q * C_DIM + h * DHEAD + d] = (_Float16)(o[j][r] / lrun[r]);
    }
}

// ---------------------------------------------------------------------------
// metric = k.mean(axis=heads), L2-normalized. one block (64 thr) per token
// ---------------------------------------------------------------------------
__global__ __launch_bounds__(64) void k_metric(
    const _Float16* __restrict__ qkv, float* __restrict__ metric, int Ntok) {
  const int n = blockIdx.x;
  const int d = threadIdx.x;
  float v = 0.f;
#pragma unroll
  for (int h = 0; h < HEADS; ++h)
    v += (float)qkv[(size_t)n * (3 * C_DIM) + C_DIM + h * DHEAD + d];
  v *= (1.f / (float)HEADS);
  float sq = v * v;
  for (int m = 16; m >= 1; m >>= 1) sq += __shfl_xor(sq, m, 32);
  __shared__ float part[2];
  if ((d & 31) == 0) part[d >> 5] = sq;
  __syncthreads();
  float tot = part[0] + part[1];
  metric[(size_t)n * DHEAD + d] = v * rsqrtf(fmaxf(tot, 1e-20f));
}

// ---------------------------------------------------------------------------
// merge scores: per even-token row, max/argmax over odd tokens
// ---------------------------------------------------------------------------
__global__ __launch_bounds__(256) void k_merge_scores(
    const float* __restrict__ metric, const float* __restrict__ thr,
    int* __restrict__ mask, int* __restrict__ nidx, int Nsrc) {
  const int n = blockIdx.x;
  const int tid = threadIdx.x;
  const float* a = metric + (size_t)(2 * n) * DHEAD;
  float best = -1e30f; int bidx = 0x7fffffff;
  for (int m = tid; m < Nsrc; m += 256) {
    const float* b = metric + (size_t)(2 * m + 1) * DHEAD;
    float dot = 0.f;
#pragma unroll
    for (int d = 0; d < DHEAD; ++d) dot += a[d] * b[d];
    if (dot > best || (dot == best && m < bidx)) { best = dot; bidx = m; }
  }
  __shared__ float bv[256];
  __shared__ int   bi[256];
  bv[tid] = best; bi[tid] = bidx;
  __syncthreads();
  for (int s = 128; s > 0; s >>= 1) {
    if (tid < s) {
      if (bv[tid + s] > bv[tid] || (bv[tid + s] == bv[tid] && bi[tid + s] < bi[tid])) {
        bv[tid] = bv[tid + s]; bi[tid] = bi[tid + s];
      }
    }
    __syncthreads();
  }
  if (tid == 0) {
    if (n == 0) { mask[0] = 0; nidx[0] = 0; }     // row 0 forced -inf in ref
    else        { mask[n] = (bv[0] > thr[0]) ? 1 : 0; nidx[n] = bi[0]; }
  }
}

// ---------------------------------------------------------------------------
// exclusive scan of !mask (single block, Nsrc <= 1024)
// ---------------------------------------------------------------------------
__global__ __launch_bounds__(1024) void k_scan(
    const int* __restrict__ mask, int* __restrict__ pos, int* __restrict__ total, int Nsrc) {
  __shared__ int buf[1024];
  const int t = threadIdx.x;
  int v = (t < Nsrc) ? (mask[t] ? 0 : 1) : 0;
  buf[t] = v;
  __syncthreads();
  for (int off = 1; off < 1024; off <<= 1) {
    int add = (t >= off) ? buf[t - off] : 0;
    __syncthreads();
    buf[t] += add;
    __syncthreads();
  }
  if (t < Nsrc) pos[t] = buf[t] - v;
  if (t == Nsrc - 1) *total = buf[t];
}

// ---------------------------------------------------------------------------
// token merge: init (dst copy + unmerged gather), scatter-add, divide
// ---------------------------------------------------------------------------
__global__ __launch_bounds__(256) void k_merge_init(
    const float* __restrict__ x1, const float* __restrict__ sz,
    const int* __restrict__ mask, const int* __restrict__ pos, const int* __restrict__ totalU,
    float* __restrict__ xm, float* __restrict__ sm, int Nsrc, int C) {
  const int b = blockIdx.x;
  const int U = *totalU;
  int srcTok, orow;
  if (b < Nsrc) { srcTok = 2 * b + 1; orow = U + b; }            // dst tokens
  else {
    const int n = b - Nsrc;
    if (mask[n]) return;                                          // merged src handled by scatter
    srcTok = 2 * n; orow = pos[n];
  }
  const float s = sz[srcTok];
  for (int c = threadIdx.x; c < C; c += 256)
    xm[(size_t)orow * C + c] = x1[(size_t)srcTok * C + c] * s;
  if (threadIdx.x == 0) sm[orow] = s;
}

__global__ __launch_bounds__(256) void k_merge_scatter(
    const float* __restrict__ x1, const float* __restrict__ sz,
    const int* __restrict__ mask, const int* __restrict__ nidx, const int* __restrict__ totalU,
    float* __restrict__ xm, float* __restrict__ sm, int C) {
  const int n = blockIdx.x;
  if (!mask[n]) return;
  const int srcTok = 2 * n;
  const int orow = *totalU + nidx[n];
  const float s = sz[srcTok];
  for (int c = threadIdx.x; c < C; c += 256)
    atomicAdd(&xm[(size_t)orow * C + c], x1[(size_t)srcTok * C + c] * s);
  if (threadIdx.x == 0) atomicAdd(&sm[orow], s);
}

__global__ __launch_bounds__(256) void k_merge_div(
    float* __restrict__ xm, const float* __restrict__ sm, int C) {
  const int r = blockIdx.x;
  const float inv = 1.f / sm[r];
  for (int c = threadIdx.x; c < C; c += 256) xm[(size_t)r * C + c] *= inv;
}

__global__ void k_copy(const float* __restrict__ in, float* __restrict__ out, int n) {
  int i = blockIdx.x * blockDim.x + threadIdx.x;
  if (i < n) out[i] = in[i];
}

// ---------------------------------------------------------------------------
// launcher
// ---------------------------------------------------------------------------
extern "C" void kernel_launch(void* const* d_in, const int* in_sizes, int n_in,
                              void* d_out, int out_size, void* d_ws, size_t ws_size,
                              hipStream_t stream) {
  const float* x      = (const float*)d_in[0];
  const float* sz     = (const float*)d_in[1];
  const float* qkv_w  = (const float*)d_in[2];
  const float* proj_w = (const float*)d_in[3];
  const float* proj_b = (const float*)d_in[4];
  const float* ln1_g  = (const float*)d_in[5];
  const float* ln1_b  = (const float*)d_in[6];
  const float* ln2_g  = (const float*)d_in[7];
  const float* ln2_b  = (const float*)d_in[8];
  const float* fc1_w  = (const float*)d_in[9];
  const float* fc1_b  = (const float*)d_in[10];
  const float* fc2_w  = (const float*)d_in[11];
  const float* fc2_b  = (const float*)d_in[12];
  const float* thr    = (const float*)d_in[13];

  const int N = in_sizes[1];              // tokens (2048)
  const int C = C_DIM;                    // 1024
  const int Nsrc = N / 2;
  const int Nprime = out_size / (C + 1);  // merged token count (x: N'*C, size: N')

  // workspace carve
  char* base = (char*)d_ws;
  size_t off = 0;
  auto alloc = [&](size_t bytes) -> void* {
    off = (off + 255) & ~(size_t)255;
    void* p = base + off;
    off += bytes;
    return p;
  };
  _Float16* h16     = (_Float16*)alloc((size_t)N * C * 2);
  _Float16* qkvwT   = (_Float16*)alloc((size_t)C * 3 * C * 2);   // [3C][C]
  _Float16* projwT  = (_Float16*)alloc((size_t)C * C * 2);       // [C][C]
  _Float16* fc1wT   = (_Float16*)alloc((size_t)C * 4 * C * 2);   // [4C][C]
  _Float16* fc2wT   = (_Float16*)alloc((size_t)4 * C * C * 2);   // [C][4C]
  _Float16* qkv16   = (_Float16*)alloc((size_t)N * 3 * C * 2);
  _Float16* attn16  = (_Float16*)alloc((size_t)N * C * 2);
  _Float16* ln216   = (_Float16*)alloc((size_t)N * C * 2);
  _Float16* fc1h16  = (_Float16*)alloc((size_t)N * 4 * C * 2);
  float*    x1      = (float*)alloc((size_t)N * C * 4);
  float*    xm      = (float*)alloc((size_t)N * C * 4);
  float*    logsz   = (float*)alloc((size_t)N * 4);
  float*    metric  = (float*)alloc((size_t)N * DHEAD * 4);
  float*    sm      = (float*)alloc((size_t)N * 4);
  int*      mask    = (int*)alloc((size_t)Nsrc * 4);
  int*      nidx    = (int*)alloc((size_t)Nsrc * 4);
  int*      pos     = (int*)alloc((size_t)Nsrc * 4);
  int*      totalU  = (int*)alloc(4);

  // 1) one-time weight f32->f16 + transpose ([K][N] -> [N][K])
  k_transpose_f16<<<dim3(3 * C / 32, C / 32),     256, 0, stream>>>(qkv_w,  qkvwT,  C,     3 * C);
  k_transpose_f16<<<dim3(C / 32,     C / 32),     256, 0, stream>>>(proj_w, projwT, C,     C);
  k_transpose_f16<<<dim3(4 * C / 32, C / 32),     256, 0, stream>>>(fc1_w,  fc1wT,  C,     4 * C);
  k_transpose_f16<<<dim3(C / 32,     4 * C / 32), 256, 0, stream>>>(fc2_w,  fc2wT,  4 * C, C);
  k_logsize<<<(N + 255) / 256, 256, 0, stream>>>(sz, logsz, N);

  // 2) LN1 -> h16
  k_layernorm_f16<<<N, 256, 0, stream>>>(x, ln1_g, ln1_b, h16, C);

  // 3) QKV GEMM (M=N, N=3C, K=C) -> f16
  k_gemm_f16<<<dim3(3 * C / TN, N / TM), 256, 0, stream>>>(
      h16, qkvwT, nullptr, nullptr, nullptr, qkv16, N, 3 * C, C, 0);

  // 4) flash attention -> attn16
  k_attn_flash<<<dim3(N / 128, HEADS), 256, 0, stream>>>(qkv16, logsz, attn16, N);

  // 5) metric from K (head mean, normalized)
  k_metric<<<N, 64, 0, stream>>>(qkv16, metric, N);

  // 6) proj GEMM + bias + residual(x) -> x1 (f32)
  k_gemm_f16<<<dim3(C / TN, N / TM), 256, 0, stream>>>(
      attn16, projwT, proj_b, x, x1, nullptr, N, C, C, 0);

  // 7) merge bookkeeping
  k_merge_scores<<<Nsrc, 256, 0, stream>>>(metric, thr, mask, nidx, Nsrc);
  k_scan<<<1, 1024, 0, stream>>>(mask, pos, totalU, Nsrc);

  // 8) build merged tokens
  k_merge_init<<<2 * Nsrc, 256, 0, stream>>>(x1, sz, mask, pos, totalU, xm, sm, Nsrc, C);
  k_merge_scatter<<<Nsrc, 256, 0, stream>>>(x1, sz, mask, nidx, totalU, xm, sm, C);
  k_merge_div<<<Nprime, 256, 0, stream>>>(xm, sm, C);

  // 9) LN2 -> ln216
  k_layernorm_f16<<<Nprime, 256, 0, stream>>>(xm, ln2_g, ln2_b, ln216, C);

  // 10) MLP: fc1 + GELU -> f16 ; fc2 + bias + residual(xm) -> d_out
  const int mt = (Nprime + TM - 1) / TM;
  k_gemm_f16<<<dim3(4 * C / TN, mt), 256, 0, stream>>>(
      ln216, fc1wT, fc1_b, nullptr, nullptr, fc1h16, Nprime, 4 * C, C, 1);
  k_gemm_f16<<<dim3(C / TN, mt), 256, 0, stream>>>(
      fc1h16, fc2wT, fc2_b, xm, (float*)d_out, nullptr, Nprime, C, 4 * C, 0);

  // 11) size output tail
  k_copy<<<(Nprime + 255) / 256, 256, 0, stream>>>(sm, (float*)d_out + (size_t)Nprime * C, Nprime);
}
